// Encoder_66760971649392
// MI455X (gfx1250) — matrix operand order repaired and checked by reading, not compiled, not run on hardware
//
#include <hip/hip_runtime.h>
#include <hip/hip_bf16.h>

// ---------------------------------------------------------------------------
// Model constants (from reference)
// ---------------------------------------------------------------------------
#define BSZ      2
#define HIMG     64
#define WIMG     4096
#define C1CH     16
#define C2CH     32
#define LSEQ     1024      // W/4
#define DMODEL   512       // C2 * H/4
#define DINNER   1024
#define DSTATE   64
#define NH       16
#define HD       64
#define CONVDIM  1152      // DINNER + 2*DSTATE
#define DINPROJ  2192      // 2*DINNER + 2*DSTATE + NH
#define FFH      1024
#define MROWS    (BSZ * LSEQ)   // 2048
#define EPSV     1e-5f

typedef __attribute__((ext_vector_type(16))) __bf16       v16bf;
typedef __attribute__((ext_vector_type(8)))  float        v8f;
typedef __attribute__((ext_vector_type(4)))  unsigned int u32x4;

static __device__ __forceinline__ int imin(int a, int b) { return a < b ? a : b; }

static __device__ __forceinline__ unsigned short f2bf(float f) {
    unsigned int u = __float_as_uint(f);
    u += 0x7FFFu + ((u >> 16) & 1u);   // round-to-nearest-even
    return (unsigned short)(u >> 16);
}

static __device__ __forceinline__ float siluf(float v) {
    return v / (1.0f + expf(-v));
}

// ---------------------------------------------------------------------------
// Async global->LDS staging (CDNA5 ASYNCcnt path), guarded so the file still
// compiles on toolchains without the gfx1250 async builtins.
// Builtin prototype (from compiler diagnostic):
//   void __builtin_amdgcn_global_load_async_to_lds_b128(v4i AS1* src,
//                                                       v4i AS3* dst,
//                                                       Imm offset, Imm cpol)
// ---------------------------------------------------------------------------
#if defined(__has_builtin)
#  if __has_builtin(__builtin_amdgcn_global_load_async_to_lds_b128) && \
      __has_builtin(__builtin_amdgcn_s_wait_asynccnt)
#    define USE_ASYNC_COPY 1
#  endif
#endif
#ifndef USE_ASYNC_COPY
#  define USE_ASYNC_COPY 0
#endif

typedef int v4i_vs __attribute__((vector_size(4 * sizeof(int))));
typedef __attribute__((address_space(1))) v4i_vs gv4i;   // global int4
typedef __attribute__((address_space(3))) v4i_vs lv4i;   // LDS int4

// Copy 16 bytes (8 bf16) global -> LDS.
static __device__ __forceinline__ void copy16_g2l(const unsigned short* g, unsigned short* l)
{
#if USE_ASYNC_COPY
    __builtin_amdgcn_global_load_async_to_lds_b128((gv4i*)g, (lv4i*)l, 0, 0);
#else
    *(u32x4*)l = *(const u32x4*)g;
#endif
}

static __device__ __forceinline__ void wait_stage_and_barrier()
{
#if USE_ASYNC_COPY
    __builtin_amdgcn_s_wait_asynccnt(0);
#endif
    __syncthreads();
}

struct FragBits { u32x4 lo, hi; };

// A fragment (16x32 bf16, MxK): lane = M; halves 0-7 -> K = k0..k0+7,
// halves 8-15 -> K = k0+16..k0+23, with k0 = 8*(lane>=16). Two b128 loads.
static __device__ __forceinline__ v16bf load_fragA(const unsigned short* p) {
    FragBits fb;
    fb.lo = *(const u32x4*)(p);
    fb.hi = *(const u32x4*)(p + 16);
    return __builtin_bit_cast(v16bf, fb);
}
// B fragment (32x16 bf16, KxN): lane = N; halves 0-7 -> K = k0..k0+7, halves 8-15 -> K = k0+16..k0+23, k0 = 8*(lane>=16).
static __device__ __forceinline__ v16bf load_fragB(const unsigned short* p) {
    FragBits fb;
    fb.lo = *(const u32x4*)(p);
    fb.hi = *(const u32x4*)(p + 16);
    return __builtin_bit_cast(v16bf, fb);
}

// ---------------------------------------------------------------------------
// WMMA GEMM: C[m,n] = sum_k A[m,k] * W[n,k] (+ bias[n])
// A: (M,K) bf16 row-major, W: (N,K) bf16 row-major, C: (M,N) f32.
// Block = 8 waves (256 thr). Wave tile 32x32 (2x2 WMMA). Block tile 64x128.
// Block tile staged to LDS once per k-step via async DMA, double-buffered;
// rows padded to 40 halves (80 B) so wave-level b128 LDS reads are
// bank-conflict-free while staged 16 B chunks stay 16 B aligned.
// ---------------------------------------------------------------------------
#define KSTEP   32
#define TSTRIDE 40          // padded row stride in halves (80 B = 5 * 16 B)

__global__ __launch_bounds__(256) void gemm_bf16_wmma_kernel(
    const unsigned short* __restrict__ A,
    const unsigned short* __restrict__ W,
    const float* __restrict__ bias,
    float* __restrict__ C,
    int M, int N, int K)
{
    __shared__ unsigned short Abuf[2][64 * TSTRIDE];     // 2 x 5 KB
    __shared__ unsigned short Bbuf[2][128 * TSTRIDE];    // 2 x 10 KB

    const int tid  = threadIdx.x;
    const int lane = tid & 31;
    const int wid  = tid >> 5;
    const int mblk = blockIdx.x * 64;
    const int nblk = blockIdx.y * 128;
    const int lm = lane & 15;
    const int lh = lane >> 4;

    // ---- staging descriptors: one 16 B A-chunk + two 16 B B-chunks / thread
    const int srow = tid >> 2;                 // 0..63
    const int scol = (tid & 3) << 3;           // halves: 0,8,16,24
    const unsigned short* gA  = A + (size_t)imin(mblk + srow,      M - 1) * K + scol;
    const unsigned short* gB0 = W + (size_t)imin(nblk + srow,      N - 1) * K + scol;
    const unsigned short* gB1 = W + (size_t)imin(nblk + 64 + srow, N - 1) * K + scol;

    auto stage = [&](int buf, int k) {
        copy16_g2l(gA  + k, &Abuf[buf][srow * TSTRIDE + scol]);
        copy16_g2l(gB0 + k, &Bbuf[buf][srow * TSTRIDE + scol]);
        copy16_g2l(gB1 + k, &Bbuf[buf][(64 + srow) * TSTRIDE + scol]);
        __builtin_prefetch(gB0 + k + 2 * KSTEP, 0, 0);   // global_prefetch_b8
    };

    // ---- compute-side LDS fragment addresses
    const int ml0 = ((wid >> 2) << 5) + lm;    // wave's A rows (local 0..63)
    const int nl0 = ((wid & 3) << 5) + lm;     // wave's B rows (local 0..127)

    v8f acc00 = {}, acc01 = {}, acc10 = {}, acc11 = {};

    stage(0, 0);
    int buf = 0;
    for (int k = 0; k < K; k += KSTEP) {
        wait_stage_and_barrier();                  // stage(buf,k) done; readers of buf^1 done
        if (k + KSTEP < K) stage(buf ^ 1, k + KSTEP);

        const unsigned short* la0 = &Abuf[buf][ ml0       * TSTRIDE + (lh << 3)];
        const unsigned short* la1 = &Abuf[buf][(ml0 + 16) * TSTRIDE + (lh << 3)];
        const unsigned short* lb0 = &Bbuf[buf][ nl0       * TSTRIDE + (lh << 3)];
        const unsigned short* lb1 = &Bbuf[buf][(nl0 + 16) * TSTRIDE + (lh << 3)];
        v16bf a0 = load_fragA(la0);
        v16bf a1 = load_fragA(la1);
        v16bf b0 = load_fragB(lb0);
        v16bf b1 = load_fragB(lb1);
        acc00 = __builtin_amdgcn_wmma_f32_16x16x32_bf16(false, a0, false, b0, (short)0, acc00, false, false);
        acc01 = __builtin_amdgcn_wmma_f32_16x16x32_bf16(false, a0, false, b1, (short)0, acc01, false, false);
        acc10 = __builtin_amdgcn_wmma_f32_16x16x32_bf16(false, a1, false, b0, (short)0, acc10, false, false);
        acc11 = __builtin_amdgcn_wmma_f32_16x16x32_bf16(false, a1, false, b1, (short)0, acc11, false, false);
        buf ^= 1;
    }

    // C/D layout: VGPR r holds M = r + 8*(lane>=16), N = lane&15.
    const int mrb = mblk + ((wid >> 2) << 5) + (lh << 3);
    const int ncA = nblk + ((wid & 3) << 5) + lm;
    const int ncB = ncA + 16;
    const float bA = (bias && ncA < N) ? bias[ncA] : 0.f;
    const float bB = (bias && ncB < N) ? bias[ncB] : 0.f;
    for (int r = 0; r < 8; ++r) {
        int mA = mrb + r, mB = mrb + 16 + r;
        if (mA < M) {
            if (ncA < N) C[(size_t)mA * N + ncA] = acc00[r] + bA;
            if (ncB < N) C[(size_t)mA * N + ncB] = acc01[r] + bB;
        }
        if (mB < M) {
            if (ncA < N) C[(size_t)mB * N + ncA] = acc10[r] + bA;
            if (ncB < N) C[(size_t)mB * N + ncB] = acc11[r] + bB;
        }
    }
}

// ---------------------------------------------------------------------------
// Conv stem
// ---------------------------------------------------------------------------
__global__ void conv1_kernel(const float* __restrict__ img, const float* __restrict__ w,
                             float* __restrict__ out)
{
    long i = (long)blockIdx.x * blockDim.x + threadIdx.x;
    const long OW = WIMG / 2, OH = HIMG / 2;
    const long total = (long)BSZ * C1CH * OH * OW;
    if (i >= total) return;
    int ow = (int)(i % OW);
    int oh = (int)((i / OW) % OH);
    int c  = (int)((i / (OW * OH)) % C1CH);
    int b  = (int)(i / (OW * OH * C1CH));
    float acc = 0.f;
    const float* ib = img + (long)b * HIMG * WIMG;
    const float* wc = w + c * 49;
    for (int kh = 0; kh < 7; ++kh) {
        int ih = oh * 2 - 3 + kh;
        if (ih < 0 || ih >= HIMG) continue;
        for (int kw = 0; kw < 7; ++kw) {
            int iw = ow * 2 - 3 + kw;
            if (iw < 0 || iw >= WIMG) continue;
            acc += ib[(long)ih * WIMG + iw] * wc[kh * 7 + kw];
        }
    }
    out[i] = acc;
}

__global__ void conv2_kernel(const float* __restrict__ x, const float* __restrict__ w,
                             float* __restrict__ out)
{
    long i = (long)blockIdx.x * blockDim.x + threadIdx.x;
    const long IH = HIMG / 2, IW = WIMG / 2;       // 32 x 2048
    const long OH = HIMG / 4, OW = WIMG / 4;       // 16 x 1024
    const long total = (long)BSZ * C2CH * OH * OW;
    if (i >= total) return;
    int ow = (int)(i % OW);
    int oh = (int)((i / OW) % OH);
    int co = (int)((i / (OW * OH)) % C2CH);
    int b  = (int)(i / (OW * OH * C2CH));
    float acc = 0.f;
    for (int ci = 0; ci < C1CH; ++ci) {
        const float* ib = x + ((long)b * C1CH + ci) * IH * IW;
        const float* wc = w + ((long)co * C1CH + ci) * 49;
        for (int kh = 0; kh < 7; ++kh) {
            int ih = oh * 2 - 3 + kh;
            if (ih < 0 || ih >= (int)IH) continue;
            for (int kw = 0; kw < 7; ++kw) {
                int iw = ow * 2 - 3 + kw;
                if (iw < 0 || iw >= (int)IW) continue;
                acc += ib[(long)ih * IW + iw] * wc[kh * 7 + kw];
            }
        }
    }
    out[i] = acc;
}

__global__ void ds_kernel(const float* __restrict__ img, const float* __restrict__ w,
                          float* __restrict__ out)
{
    long i = (long)blockIdx.x * blockDim.x + threadIdx.x;
    const long OH = HIMG / 4, OW = WIMG / 4;
    const long total = (long)BSZ * C2CH * OH * OW;
    if (i >= total) return;
    int ow = (int)(i % OW);
    int oh = (int)((i / OW) % OH);
    int c  = (int)((i / (OW * OH)) % C2CH);
    int b  = (int)(i / (OW * OH * C2CH));
    out[i] = img[(long)b * HIMG * WIMG + (long)(oh * 4) * WIMG + (ow * 4)] * w[c];
}

// ---------------------------------------------------------------------------
// Batch-norm statistics: one block per channel; generic strides.
// ---------------------------------------------------------------------------
__global__ void bn_stats_kernel(const float* __restrict__ x, float* __restrict__ mean,
                                float* __restrict__ rstd, int Bn, long S,
                                long sB, long sC, long sS)
{
    int c = blockIdx.x;
    int tid = threadIdx.x;
    long n = (long)Bn * S;
    float s = 0.f, ss = 0.f;
    for (long i = tid; i < n; i += blockDim.x) {
        long b = i / S, sp = i - b * S;
        float v = x[b * sB + (long)c * sC + sp * sS];
        s += v; ss += v * v;
    }
    __shared__ float sh0[256], sh1[256];
    sh0[tid] = s; sh1[tid] = ss;
    __syncthreads();
    for (int o = 128; o > 0; o >>= 1) {
        if (tid < o) { sh0[tid] += sh0[tid + o]; sh1[tid] += sh1[tid + o]; }
        __syncthreads();
    }
    if (tid == 0) {
        float m = sh0[0] / (float)n;
        float v = sh1[0] / (float)n - m * m;
        mean[c] = m;
        rstd[c] = rsqrtf(v + EPSV);
    }
}

// In-place BN for (B,C,S) layout.
__global__ void bn_apply_kernel(float* __restrict__ x, const float* __restrict__ mean,
                                const float* __restrict__ rstd, const float* __restrict__ g,
                                const float* __restrict__ bb, int C, long S, long total)
{
    long i = (long)blockIdx.x * blockDim.x + threadIdx.x;
    if (i >= total) return;
    int c = (int)((i / S) % C);
    x[i] = (x[i] - mean[c]) * rstd[c] * g[c] + bb[c];
}

// tokens = relu(bnA(t2) + bnB(t3)), (B,C,S) layout.
__global__ void stem_fuse_kernel(const float* __restrict__ t2, const float* __restrict__ mA,
                                 const float* __restrict__ rA, const float* __restrict__ gA,
                                 const float* __restrict__ bA,
                                 const float* __restrict__ t3, const float* __restrict__ mB,
                                 const float* __restrict__ rB, const float* __restrict__ gB,
                                 const float* __restrict__ bB,
                                 float* __restrict__ tok, int C, long S, long total)
{
    long i = (long)blockIdx.x * blockDim.x + threadIdx.x;
    if (i >= total) return;
    int c = (int)((i / S) % C);
    float a = (t2[i] - mA[c]) * rA[c] * gA[c] + bA[c];
    float r = (t3[i] - mB[c]) * rB[c] * gB[c] + bB[c];
    float v = a + r;
    tok[i] = v > 0.f ? v : 0.f;
}

// ---------------------------------------------------------------------------
// Layout/precision movers
// ---------------------------------------------------------------------------
// (B,C,L) f32 -> (B*L, C) bf16, LDS-tiled transpose.
__global__ void transpose_bf16_kernel(const float* __restrict__ in,
                                      unsigned short* __restrict__ out, int C, int L)
{
    __shared__ float tile[32][33];
    int b  = blockIdx.z;
    int c0 = blockIdx.y * 32;
    int l0 = blockIdx.x * 32;
    for (int r = threadIdx.y; r < 32; r += 8) {
        tile[r][threadIdx.x] = in[((long)b * C + c0 + r) * L + l0 + threadIdx.x];
    }
    __syncthreads();
    for (int r = threadIdx.y; r < 32; r += 8) {
        out[((long)b * L + l0 + r) * C + c0 + threadIdx.x] = f2bf(tile[threadIdx.x][r]);
    }
}

__global__ void f32_to_bf16_kernel(const float* __restrict__ in,
                                   unsigned short* __restrict__ out, long n)
{
    long i = (long)blockIdx.x * blockDim.x + threadIdx.x;
    if (i < n) out[i] = f2bf(in[i]);
}

__global__ void copy_kernel(const float* __restrict__ in, float* __restrict__ out, long n)
{
    long i = (long)blockIdx.x * blockDim.x + threadIdx.x;
    if (i < n) out[i] = in[i];
}

// ---------------------------------------------------------------------------
// Mamba pieces
// ---------------------------------------------------------------------------
// dt = softplus(raw_dt + dt_bias); dA = exp(-exp(A_log) * dt).
__global__ void dt_kernel(const float* __restrict__ zx, const float* __restrict__ dt_bias,
                          const float* __restrict__ A_log, float* __restrict__ dt,
                          float* __restrict__ dA, long rows)
{
    long i = (long)blockIdx.x * blockDim.x + threadIdx.x;
    if (i >= rows * NH) return;
    int h = (int)(i % NH);
    long r = i / NH;
    float d = zx[r * DINPROJ + 2 * DINNER + 2 * DSTATE + h] + dt_bias[h];
    float sp = (d > 20.f) ? d : log1pf(expf(d));
    dt[i] = sp;
    dA[i] = expf(-expf(A_log[h]) * sp);
}

// Causal depthwise conv (D_CONV=4) over xBC channels + SiLU.
__global__ void dwconv_kernel(const float* __restrict__ zx, const float* __restrict__ cw,
                              const float* __restrict__ cb, float* __restrict__ xc, long rows)
{
    long i = (long)blockIdx.x * blockDim.x + threadIdx.x;
    if (i >= rows * CONVDIM) return;
    int c = (int)(i % CONVDIM);
    long r = i / CONVDIM;
    long l = r % LSEQ;
    long base = r - l;                       // start row of this batch
    float acc = cb[c];
    for (int k = 0; k < 4; ++k) {
        long lk = l + k - 3;
        if (lk >= 0)
            acc += zx[(base + lk) * DINPROJ + DINNER + c] * cw[c * 4 + k];
    }
    xc[i] = siluf(acc);
}

// Selective scan: one block per (batch, head); 64x64 state in registers
// (16 floats / thread), x/B/C staged in LDS per step, LDS partial-sum for y.
__global__ __launch_bounds__(256) void scan_kernel(
    const float* __restrict__ xc, const float* __restrict__ dt,
    const float* __restrict__ dA, const float* __restrict__ Dp,
    float* __restrict__ y)
{
    int b = blockIdx.x >> 4;
    int h = blockIdx.x & 15;
    int tid = threadIdx.x;
    int p  = tid >> 2;
    int g  = tid & 3;
    int n0 = g * 16;
    float hs[16];
    #pragma unroll
    for (int j = 0; j < 16; ++j) hs[j] = 0.f;

    __shared__ float sx[64], sB[64], sC[64], part[256], ssc[2];
    const float Dv = Dp[h];

    for (int l = 0; l < LSEQ; ++l) {
        long row = (long)b * LSEQ + l;
        if (tid < 64) {
            sx[tid] = xc[row * CONVDIM + h * HD + tid];
            sB[tid] = xc[row * CONVDIM + DINNER + tid];
            sC[tid] = xc[row * CONVDIM + DINNER + DSTATE + tid];
        }
        if (tid == 0) {
            ssc[0] = dA[row * NH + h];
            ssc[1] = dt[row * NH + h];
        }
        __syncthreads();
        float a  = ssc[0];
        float xd = sx[p] * ssc[1];
        float accp = 0.f;
        #pragma unroll
        for (int j = 0; j < 16; ++j) {
            float hv = a * hs[j] + xd * sB[n0 + j];
            hs[j] = hv;
            accp += hv * sC[n0 + j];
        }
        part[tid] = accp;
        __syncthreads();
        if (g == 0) {
            float yv = part[tid] + part[tid + 1] + part[tid + 2] + part[tid + 3]
                     + sx[p] * Dv;
            y[row * DINNER + h * HD + p] = yv;
        }
        __syncthreads();
    }
}

// y2 = y * silu(z); RMS-norm over 1024 dims; scale by norm_w; emit bf16.
__global__ __launch_bounds__(256) void gate_rms_kernel(
    const float* __restrict__ y, const float* __restrict__ zx,
    const float* __restrict__ nw, unsigned short* __restrict__ out)
{
    long r = blockIdx.x;               // row in [0, B*L)
    int tid = threadIdx.x;
    __shared__ float sh[256];
    float vals[4];
    float ss = 0.f;
    #pragma unroll
    for (int j = 0; j < 4; ++j) {
        int c = tid + j * 256;
        float z = zx[r * DINPROJ + c];
        float v = y[r * DINNER + c] * siluf(z);
        vals[j] = v;
        ss += v * v;
    }
    sh[tid] = ss;
    __syncthreads();
    for (int o = 128; o > 0; o >>= 1) {
        if (tid < o) sh[tid] += sh[tid + o];
        __syncthreads();
    }
    float rst = rsqrtf(sh[0] / (float)DINNER + EPSV);
    #pragma unroll
    for (int j = 0; j < 4; ++j) {
        int c = tid + j * 256;
        out[r * DINNER + c] = f2bf(vals[j] * rst * nw[c]);
    }
}

// tokens(B,C,L) += bn(proj(B*L,C))
__global__ void bn_res_kernel(const float* __restrict__ proj, const float* __restrict__ mean,
                              const float* __restrict__ rstd, const float* __restrict__ g,
                              const float* __restrict__ bb, float* __restrict__ tok,
                              int C, int L, long total)
{
    long i = (long)blockIdx.x * blockDim.x + threadIdx.x;
    if (i >= total) return;
    int l = (int)(i % L);
    int c = (int)((i / L) % C);
    int b = (int)(i / ((long)L * C));
    float v = proj[((long)b * L + l) * C + c];
    v = (v - mean[c]) * rstd[c] * g[c] + bb[c];
    tok[i] += v;
}

// (B*L,C) f32 -> relu(bn(.)) -> bf16 same layout.
__global__ void bn_relu_bf16_kernel(const float* __restrict__ F, const float* __restrict__ mean,
                                    const float* __restrict__ rstd, const float* __restrict__ g,
                                    const float* __restrict__ bb,
                                    unsigned short* __restrict__ out, int C, long total)
{
    long i = (long)blockIdx.x * blockDim.x + threadIdx.x;
    if (i >= total) return;
    int c = (int)(i % C);
    float v = (F[i] - mean[c]) * rstd[c] * g[c] + bb[c];
    out[i] = f2bf(v > 0.f ? v : 0.f);
}

// ---------------------------------------------------------------------------
// Host orchestration
// ---------------------------------------------------------------------------
static inline long ceil_div(long a, long b) { return (a + b - 1) / b; }

extern "C" void kernel_launch(void* const* d_in, const int* in_sizes, int n_in,
                              void* d_out, int out_size, void* d_ws, size_t ws_size,
                              hipStream_t stream)
{
    (void)in_sizes; (void)n_in; (void)out_size; (void)ws_size;

    const float* image   = (const float*)d_in[0];
    const float* conv1_w = (const float*)d_in[1];
    const float* bn1_g   = (const float*)d_in[2];
    const float* bn1_b   = (const float*)d_in[3];
    const float* conv2_w = (const float*)d_in[4];
    const float* bn2_g   = (const float*)d_in[5];
    const float* bn2_b   = (const float*)d_in[6];
    const float* ds_w    = (const float*)d_in[7];
    const float* ds_g    = (const float*)d_in[8];
    const float* ds_b    = (const float*)d_in[9];
    const float* blk_inproj_w  = (const float*)d_in[10];
    const float* blk_conv_w    = (const float*)d_in[11];
    const float* blk_conv_b    = (const float*)d_in[12];
    const float* blk_dt_bias   = (const float*)d_in[13];
    const float* blk_A_log     = (const float*)d_in[14];
    const float* blk_D         = (const float*)d_in[15];
    const float* blk_norm_w    = (const float*)d_in[16];
    const float* blk_outproj_w = (const float*)d_in[17];
    const float* blk_bn_g      = (const float*)d_in[18];
    const float* blk_bn_b      = (const float*)d_in[19];
    const float* blk_ff_w1     = (const float*)d_in[20];
    const float* blk_ff_b1     = (const float*)d_in[21];
    const float* blk_ff_bn1_g  = (const float*)d_in[22];
    const float* blk_ff_bn1_b  = (const float*)d_in[23];
    const float* blk_ff_w2     = (const float*)d_in[24];
    const float* blk_ff_b2     = (const float*)d_in[25];
    const float* blk_ff_bn2_g  = (const float*)d_in[26];
    const float* blk_ff_bn2_b  = (const float*)d_in[27];

    // ---- workspace bump allocator (256B aligned) ----
    size_t off = 0;
    char* base = (char*)d_ws;
    auto alloc = [&](size_t bytes) -> void* {
        void* p = base + off;
        off = (off + bytes + 255) & ~(size_t)255;
        return p;
    };

    float* t1     = (float*)alloc((size_t)BSZ * C1CH * (HIMG/2) * (WIMG/2) * 4);   // conv1 out
    float* t2     = (float*)alloc((size_t)BSZ * C2CH * (HIMG/4) * (WIMG/4) * 4);   // conv2 out
    float* t3     = (float*)alloc((size_t)BSZ * C2CH * (HIMG/4) * (WIMG/4) * 4);   // ds out
    float* tokens = (float*)alloc((size_t)BSZ * DMODEL * LSEQ * 4);                // (B,C,L)
    float* statM  = (float*)alloc(4096 * 4);
    float* statR  = (float*)alloc(4096 * 4);
    float* statM2 = (float*)alloc(4096 * 4);
    float* statR2 = (float*)alloc(4096 * 4);
    unsigned short* tokbf = (unsigned short*)alloc((size_t)MROWS * DMODEL * 2);
    unsigned short* wbf   = (unsigned short*)alloc((size_t)DINPROJ * DMODEL * 2);  // max weight
    float* zx     = (float*)alloc((size_t)MROWS * DINPROJ * 4);
    float* dtb    = (float*)alloc((size_t)MROWS * NH * 4);
    float* dAb    = (float*)alloc((size_t)MROWS * NH * 4);
    float* xc     = (float*)alloc((size_t)MROWS * CONVDIM * 4);
    float* ybuf   = (float*)alloc((size_t)MROWS * DINNER * 4);
    unsigned short* ybf = (unsigned short*)alloc((size_t)MROWS * DINNER * 2);
    float* proj   = (float*)alloc((size_t)MROWS * DMODEL * 4);
    float* Fbuf   = (float*)alloc((size_t)MROWS * FFH * 4);
    unsigned short* Fbf = (unsigned short*)alloc((size_t)MROWS * FFH * 2);
    float* Gbuf   = (float*)alloc((size_t)MROWS * DMODEL * 4);

    const int TPB = 256;

    // ---- Conv stem ----
    {
        long n1 = (long)BSZ * C1CH * (HIMG/2) * (WIMG/2);
        conv1_kernel<<<(unsigned)ceil_div(n1, TPB), TPB, 0, stream>>>(image, conv1_w, t1);
        long S1 = (long)(HIMG/2) * (WIMG/2);
        bn_stats_kernel<<<C1CH, TPB, 0, stream>>>(t1, statM, statR, BSZ, S1,
                                                  (long)C1CH * S1, S1, 1L);
        bn_apply_kernel<<<(unsigned)ceil_div(n1, TPB), TPB, 0, stream>>>(
            t1, statM, statR, bn1_g, bn1_b, C1CH, S1, n1);

        long n2 = (long)BSZ * C2CH * (HIMG/4) * (WIMG/4);
        conv2_kernel<<<(unsigned)ceil_div(n2, TPB), TPB, 0, stream>>>(t1, conv2_w, t2);
        ds_kernel<<<(unsigned)ceil_div(n2, TPB), TPB, 0, stream>>>(image, ds_w, t3);
        long S2 = (long)(HIMG/4) * (WIMG/4);
        bn_stats_kernel<<<C2CH, TPB, 0, stream>>>(t2, statM, statR, BSZ, S2,
                                                  (long)C2CH * S2, S2, 1L);
        bn_stats_kernel<<<C2CH, TPB, 0, stream>>>(t3, statM2, statR2, BSZ, S2,
                                                  (long)C2CH * S2, S2, 1L);
        stem_fuse_kernel<<<(unsigned)ceil_div(n2, TPB), TPB, 0, stream>>>(
            t2, statM, statR, bn2_g, bn2_b,
            t3, statM2, statR2, ds_g, ds_b,
            tokens, C2CH, S2, n2);
        // tokens now holds (B, 32*16=512, 1024) contiguous == (B, DMODEL, L)
    }

    dim3 trG(LSEQ / 32, DMODEL / 32, BSZ);
    dim3 trB(32, 8);
    const long ntok = (long)BSZ * DMODEL * LSEQ;

    for (int i = 0; i < 2; ++i) {
        const float* w_in   = blk_inproj_w  + (size_t)i * DINPROJ * DMODEL;
        const float* cw     = blk_conv_w    + (size_t)i * CONVDIM * 4;
        const float* cb     = blk_conv_b    + (size_t)i * CONVDIM;
        const float* dtbias = blk_dt_bias   + (size_t)i * NH;
        const float* Alog   = blk_A_log     + (size_t)i * NH;
        const float* Dp     = blk_D         + (size_t)i * NH;
        const float* nw     = blk_norm_w    + (size_t)i * DINNER;
        const float* w_out  = blk_outproj_w + (size_t)i * DMODEL * DINNER;
        const float* bng    = blk_bn_g      + (size_t)i * DMODEL;
        const float* bnb    = blk_bn_b      + (size_t)i * DMODEL;
        const float* w1     = blk_ff_w1     + (size_t)i * FFH * DMODEL;
        const float* b1     = blk_ff_b1     + (size_t)i * FFH;
        const float* fbn1g  = blk_ff_bn1_g  + (size_t)i * FFH;
        const float* fbn1b  = blk_ff_bn1_b  + (size_t)i * FFH;
        const float* w2     = blk_ff_w2     + (size_t)i * DMODEL * FFH;
        const float* b2     = blk_ff_b2     + (size_t)i * DMODEL;
        const float* fbn2g  = blk_ff_bn2_g  + (size_t)i * DMODEL;
        const float* fbn2b  = blk_ff_bn2_b  + (size_t)i * DMODEL;

        // -- Mamba2 --
        transpose_bf16_kernel<<<trG, trB, 0, stream>>>(tokens, tokbf, DMODEL, LSEQ);
        long nwin = (long)DINPROJ * DMODEL;
        f32_to_bf16_kernel<<<(unsigned)ceil_div(nwin, TPB), TPB, 0, stream>>>(w_in, wbf, nwin);
        gemm_bf16_wmma_kernel<<<dim3(MROWS / 64, (DINPROJ + 127) / 128), 256, 0, stream>>>(
            tokbf, wbf, nullptr, zx, MROWS, DINPROJ, DMODEL);

        long ndt = (long)MROWS * NH;
        dt_kernel<<<(unsigned)ceil_div(ndt, TPB), TPB, 0, stream>>>(zx, dtbias, Alog, dtb, dAb, MROWS);
        long nxc = (long)MROWS * CONVDIM;
        dwconv_kernel<<<(unsigned)ceil_div(nxc, TPB), TPB, 0, stream>>>(zx, cw, cb, xc, MROWS);

        scan_kernel<<<BSZ * NH, 256, 0, stream>>>(xc, dtb, dAb, Dp, ybuf);
        gate_rms_kernel<<<MROWS, 256, 0, stream>>>(ybuf, zx, nw, ybf);

        long nwo = (long)DMODEL * DINNER;
        f32_to_bf16_kernel<<<(unsigned)ceil_div(nwo, TPB), TPB, 0, stream>>>(w_out, wbf, nwo);
        gemm_bf16_wmma_kernel<<<dim3(MROWS / 64, (DMODEL + 127) / 128), 256, 0, stream>>>(
            ybf, wbf, nullptr, proj, MROWS, DMODEL, DINNER);

        bn_stats_kernel<<<DMODEL, TPB, 0, stream>>>(proj, statM, statR, BSZ, (long)LSEQ,
                                                    (long)LSEQ * DMODEL, 1L, (long)DMODEL);
        bn_res_kernel<<<(unsigned)ceil_div(ntok, TPB), TPB, 0, stream>>>(
            proj, statM, statR, bng, bnb, tokens, DMODEL, LSEQ, ntok);

        // -- Feed-forward --
        transpose_bf16_kernel<<<trG, trB, 0, stream>>>(tokens, tokbf, DMODEL, LSEQ);
        long nw1 = (long)FFH * DMODEL;
        f32_to_bf16_kernel<<<(unsigned)ceil_div(nw1, TPB), TPB, 0, stream>>>(w1, wbf, nw1);
        gemm_bf16_wmma_kernel<<<dim3(MROWS / 64, (FFH + 127) / 128), 256, 0, stream>>>(
            tokbf, wbf, b1, Fbuf, MROWS, FFH, DMODEL);

        bn_stats_kernel<<<FFH, TPB, 0, stream>>>(Fbuf, statM, statR, BSZ, (long)LSEQ,
                                                 (long)LSEQ * FFH, 1L, (long)FFH);
        long nF = (long)MROWS * FFH;
        bn_relu_bf16_kernel<<<(unsigned)ceil_div(nF, TPB), TPB, 0, stream>>>(
            Fbuf, statM, statR, fbn1g, fbn1b, Fbf, FFH, nF);

        long nw2 = (long)DMODEL * FFH;
        f32_to_bf16_kernel<<<(unsigned)ceil_div(nw2, TPB), TPB, 0, stream>>>(w2, wbf, nw2);
        gemm_bf16_wmma_kernel<<<dim3(MROWS / 64, (DMODEL + 127) / 128), 256, 0, stream>>>(
            Fbf, wbf, b2, Gbuf, MROWS, DMODEL, FFH);

        bn_stats_kernel<<<DMODEL, TPB, 0, stream>>>(Gbuf, statM, statR, BSZ, (long)LSEQ,
                                                    (long)LSEQ * DMODEL, 1L, (long)DMODEL);
        bn_res_kernel<<<(unsigned)ceil_div(ntok, TPB), TPB, 0, stream>>>(
            Gbuf, statM, statR, fbn2g, fbn2b, tokens, DMODEL, LSEQ, ntok);
    }

    copy_kernel<<<(unsigned)ceil_div(ntok, TPB), TPB, 0, stream>>>(tokens, (float*)d_out, ntok);
}
